// DecoderRNN_37460704756364
// MI455X (gfx1250) — matrix operand order repaired and checked by reading, not compiled, not run on hardware
//
#include <hip/hip_runtime.h>
#include <hip/hip_bf16.h>

#define EMBED  512
#define HIDDEN 1024
#define VOCAB  32000
#define BATCH  64
#define TSEQ   64
#define GATES  (4 * HIDDEN)   // 4096

typedef __attribute__((ext_vector_type(16))) __bf16 v16bf;
typedef __attribute__((ext_vector_type(8)))  float  v8f;

union FragAB { v16bf v; unsigned int u[8]; };

__device__ __forceinline__ unsigned short f2bf(float x) {
    unsigned int u = __float_as_uint(x);
    unsigned int r = u + 0x7FFFu + ((u >> 16) & 1u);   // round-to-nearest-even
    return (unsigned short)(r >> 16);
}

// A fragment: 16x32 bf16 tile (M x K). Abase points at element [0,0]; lda in elements.
// Layout: lane holds row m=lane%16; lanes 0-15: K=0..15, lanes 16-31: K=16..31;
// VGPR j holds K pair (2*j) within the lane's 16-wide K half.
__device__ __forceinline__ void load_a(FragAB& f, const unsigned short* Abase, int lda, int lane) {
    const int m     = lane & 15;
    const int khalf = (lane >> 4) << 4;                // 0 or 16
    const unsigned short* row = Abase + (size_t)m * lda + khalf;
#pragma unroll
    for (int j = 0; j < 8; ++j) {
        const int k0 = j << 1;
        f.u[j] = *reinterpret_cast<const unsigned int*>(row + k0);
    }
}

// B fragment: 32x16 bf16 tile (K x N), sourced from (N, K) row-major memory so the
// per-lane K pairs are contiguous dwords. lanes 0-15: K=0..15, lanes 16-31: K=16..31.
__device__ __forceinline__ void load_b(FragAB& f, const unsigned short* Bt, int ldb, int lane) {
    const int n    = lane & 15;
    const int kofs = (lane >> 4) << 4;                 // 0 or 16
    const unsigned short* row = Bt + (size_t)n * ldb + kofs;
#pragma unroll
    for (int j = 0; j < 8; ++j)
        f.u[j] = *reinterpret_cast<const unsigned int*>(row + (j << 1));
}

__device__ __forceinline__ v8f wmma_bf16(const FragAB& a, const FragAB& b, v8f c) {
    return __builtin_amdgcn_wmma_f32_16x16x32_bf16(false, a.v, false, b.v, (short)0, c, false, false);
}

// ---------------- pack kernels (f32 -> bf16, with the layouts the GEMMs want) --------

__global__ void pack_x_kernel(const float* __restrict__ feat, const int* __restrict__ caps,
                              const float* __restrict__ emb, unsigned short* __restrict__ xbf) {
    int idx = blockIdx.x * 256 + threadIdx.x;          // over T*B*E
    int e  = idx & (EMBED - 1);
    int bt = idx >> 9;                                 // t*64 + b
    int b  = bt & (BATCH - 1);
    int t  = bt >> 6;
    float v;
    if (t == 0) v = feat[b * EMBED + e];
    else        v = emb[(size_t)caps[b * TSEQ + (t - 1)] * EMBED + e];
    xbf[idx] = f2bf(v);
}

__global__ void pack_w_kernel(const float* __restrict__ Wi, const float* __restrict__ Wf,
                              const float* __restrict__ Wg, const float* __restrict__ Wo,
                              unsigned short* __restrict__ WT) {
    int idx = blockIdx.x * 256 + threadIdx.x;          // over 4096*512, WT[g][e]
    int e = idx & (EMBED - 1);
    int g = idx >> 9;
    int sel = g >> 10, gl = g & (HIDDEN - 1);
    const float* src = (sel == 0) ? Wi : (sel == 1) ? Wf : (sel == 2) ? Wg : Wo;
    WT[idx] = f2bf(src[(size_t)e * HIDDEN + gl]);
}

__global__ void pack_u_kernel(const float* __restrict__ Ui, const float* __restrict__ Uf,
                              const float* __restrict__ Ug, const float* __restrict__ Uo,
                              unsigned short* __restrict__ UT) {
    int idx = blockIdx.x * 256 + threadIdx.x;          // over 4096*1024, UT[g][k]
    int k = idx & (HIDDEN - 1);
    int g = idx >> 10;
    int sel = g >> 10, gl = g & (HIDDEN - 1);
    const float* src = (sel == 0) ? Ui : (sel == 1) ? Uf : (sel == 2) ? Ug : Uo;
    UT[idx] = f2bf(src[(size_t)k * HIDDEN + gl]);
}

__global__ void pack_bias_kernel(const float* __restrict__ bi, const float* __restrict__ bf_,
                                 const float* __restrict__ bg, const float* __restrict__ bo,
                                 float* __restrict__ bc) {
    int idx = blockIdx.x * 256 + threadIdx.x;          // over 4096
    int sel = idx >> 10;
    const float* src = (sel == 0) ? bi : (sel == 1) ? bf_ : (sel == 2) ? bg : bo;
    bc[idx] = src[idx & (HIDDEN - 1)];
}

__global__ void pack_lw_kernel(const float* __restrict__ lw, unsigned short* __restrict__ lwbf) {
    int idx = blockIdx.x * 256 + threadIdx.x;          // over V*H, same (V,K) layout
    lwbf[idx] = f2bf(lw[idx]);
}

__global__ void init_state_kernel(unsigned short* __restrict__ hbf, float* __restrict__ cst) {
    int idx = blockIdx.x * 256 + threadIdx.x;          // over 64*1024
    hbf[idx] = 0;
    cst[idx] = 0.0f;
}

// ---------------- LSTM step: gates = [x_t ; h] @ [W ; U] + b  (M=64, N=4096, K=1536) ---

__global__ void lstm_gates_kernel(const unsigned short* __restrict__ xbf,
                                  const unsigned short* __restrict__ WT,
                                  const unsigned short* __restrict__ UT,
                                  const unsigned short* __restrict__ hbf,
                                  const float* __restrict__ bc,
                                  float* __restrict__ gates, int t) {
    const int lane   = threadIdx.x & 31;
    const int wave   = threadIdx.x >> 5;
    const int w      = blockIdx.x * 8 + wave;          // 256 waves total
    const int m_tile = w & 3;                          // M = 64 -> 4 tiles
    const int n_base = (w >> 2) * 64;                  // N = 4096 -> 64 quads

    v8f acc0 = {}, acc1 = {}, acc2 = {}, acc3 = {};

    // Phase 1: x_t @ W, K = 512
    const unsigned short* Abase = xbf + (size_t)(t * BATCH + m_tile * 16) * EMBED;
    for (int kk = 0; kk < EMBED; kk += 32) {
        FragAB a; load_a(a, Abase + kk, EMBED, lane);
        FragAB b0; load_b(b0, WT + (size_t)(n_base +  0) * EMBED + kk, EMBED, lane);
        acc0 = wmma_bf16(a, b0, acc0);
        FragAB b1; load_b(b1, WT + (size_t)(n_base + 16) * EMBED + kk, EMBED, lane);
        acc1 = wmma_bf16(a, b1, acc1);
        FragAB b2; load_b(b2, WT + (size_t)(n_base + 32) * EMBED + kk, EMBED, lane);
        acc2 = wmma_bf16(a, b2, acc2);
        FragAB b3; load_b(b3, WT + (size_t)(n_base + 48) * EMBED + kk, EMBED, lane);
        acc3 = wmma_bf16(a, b3, acc3);
    }
    // Phase 2: h @ U, K = 1024
    const unsigned short* Hbase = hbf + (size_t)(m_tile * 16) * HIDDEN;
    for (int kk = 0; kk < HIDDEN; kk += 32) {
        FragAB a; load_a(a, Hbase + kk, HIDDEN, lane);
        FragAB b0; load_b(b0, UT + (size_t)(n_base +  0) * HIDDEN + kk, HIDDEN, lane);
        acc0 = wmma_bf16(a, b0, acc0);
        FragAB b1; load_b(b1, UT + (size_t)(n_base + 16) * HIDDEN + kk, HIDDEN, lane);
        acc1 = wmma_bf16(a, b1, acc1);
        FragAB b2; load_b(b2, UT + (size_t)(n_base + 32) * HIDDEN + kk, HIDDEN, lane);
        acc2 = wmma_bf16(a, b2, acc2);
        FragAB b3; load_b(b3, UT + (size_t)(n_base + 48) * HIDDEN + kk, HIDDEN, lane);
        acc3 = wmma_bf16(a, b3, acc3);
    }

    const int n = n_base + (lane & 15);
#pragma unroll
    for (int r = 0; r < 8; ++r) {
        int m = m_tile * 16 + r + ((lane >> 4) << 3);
        float* row = gates + (size_t)m * GATES;
        row[n     ] = acc0[r] + bc[n     ];
        row[n + 16] = acc1[r] + bc[n + 16];
        row[n + 32] = acc2[r] + bc[n + 32];
        row[n + 48] = acc3[r] + bc[n + 48];
    }
}

__global__ void lstm_cell_kernel(const float* __restrict__ gates,
                                 float* __restrict__ cst,
                                 unsigned short* __restrict__ hbf,
                                 unsigned short* __restrict__ hsbf, int t) {
    int idx = blockIdx.x * 256 + threadIdx.x;          // over 64*1024
    int b = idx >> 10;
    int k = idx & (HIDDEN - 1);
    const float* g = gates + (size_t)b * GATES;
    float i_t = 1.0f / (1.0f + expf(-g[k]));
    float f_t = 1.0f / (1.0f + expf(-g[HIDDEN + k]));
    float g_t = tanhf(g[2 * HIDDEN + k]);
    float o_t = 1.0f / (1.0f + expf(-g[3 * HIDDEN + k]));
    float c   = f_t * cst[idx] + i_t * g_t;
    cst[idx]  = c;
    float h   = o_t * tanhf(c);
    unsigned short hb = f2bf(h);
    hbf[idx] = hb;
    hsbf[(size_t)(t * BATCH + b) * HIDDEN + k] = hb;
}

// ---------------- logits: (T*B, H) @ (H, V) -> scatter to (B, T, V) --------------------
//
// Block = 256 threads (8 waves). Block tile: M=256 x N=64.
// Each wave computes a 32x64 tile (2 M-subtiles x 4 N-subtiles = 8 accumulators), so
// every B fragment staged in LDS feeds two WMMAs. The 64x32 bf16 B tile (4 KB) is
// staged cooperatively into LDS (one b128 per thread), double-buffered, one barrier
// per k-chunk. B global traffic drops 8x vs per-wave loading.

__global__ void __launch_bounds__(256)
logits_kernel(const unsigned short* __restrict__ hsbf,
              const unsigned short* __restrict__ lwbf,
              const float* __restrict__ lb,
              float* __restrict__ out) {
    __shared__ __align__(16) unsigned short Bsm[2][64 * 32];   // [buf][n][k], 4 KB each

    const int tid    = threadIdx.x;
    const int lane   = tid & 31;
    const int wave   = tid >> 5;
    const int m_base = (blockIdx.y * 8 + wave) * 32;           // wave rows m_base..m_base+31
    const int n_base = blockIdx.x * 64;

    // cooperative B stage: thread -> row n = tid/4, 16 bytes at k-offset (tid%4)*8
    const int sn = tid >> 2;
    const int sk = (tid & 3) << 3;
    const unsigned short* bsrc = lwbf + (size_t)(n_base + sn) * HIDDEN + sk;

    v8f accA0 = {}, accA1 = {}, accA2 = {}, accA3 = {};        // rows m_base..+15
    v8f accB0 = {}, accB1 = {}, accB2 = {}, accB3 = {};        // rows m_base+16..+31

    const unsigned short* Abase0 = hsbf + (size_t)m_base * HIDDEN;
    const unsigned short* Abase1 = Abase0 + 16 * HIDDEN;

    // prologue: stage chunk 0
    *reinterpret_cast<uint4*>(&Bsm[0][sn * 32 + sk]) = *reinterpret_cast<const uint4*>(bsrc);
    __syncthreads();

    for (int kc = 0; kc < HIDDEN / 32; ++kc) {
        const int cur = kc & 1;
        const int kk  = kc * 32;
        if (kc + 1 < HIDDEN / 32) {                            // stage next chunk
            *reinterpret_cast<uint4*>(&Bsm[cur ^ 1][sn * 32 + sk]) =
                *reinterpret_cast<const uint4*>(bsrc + (kk + 32));
        }

        FragAB a0; load_a(a0, Abase0 + kk, HIDDEN, lane);
        FragAB a1; load_a(a1, Abase1 + kk, HIDDEN, lane);

        FragAB b0; load_b(b0, &Bsm[cur][ 0 * 32], 32, lane);
        accA0 = wmma_bf16(a0, b0, accA0);
        accB0 = wmma_bf16(a1, b0, accB0);
        FragAB b1; load_b(b1, &Bsm[cur][16 * 32], 32, lane);
        accA1 = wmma_bf16(a0, b1, accA1);
        accB1 = wmma_bf16(a1, b1, accB1);
        FragAB b2; load_b(b2, &Bsm[cur][32 * 32], 32, lane);
        accA2 = wmma_bf16(a0, b2, accA2);
        accB2 = wmma_bf16(a1, b2, accB2);
        FragAB b3; load_b(b3, &Bsm[cur][48 * 32], 32, lane);
        accA3 = wmma_bf16(a0, b3, accA3);
        accB3 = wmma_bf16(a1, b3, accB3);

        __syncthreads();                                       // next stage / reuse safe
    }

    const int v0 = n_base + (lane & 15);
    const float bias0 = lb[v0], bias1 = lb[v0 + 16], bias2 = lb[v0 + 32], bias3 = lb[v0 + 48];
#pragma unroll
    for (int r = 0; r < 8; ++r) {
        const int rr = r + ((lane >> 4) << 3);
        {   // M-subtile 0
            int m  = m_base + rr;
            int tt = m >> 6;                                   // m = t*64 + b
            int bb = m & 63;
            float* row = out + (size_t)(bb * TSEQ + tt) * VOCAB;
            row[v0     ] = accA0[r] + bias0;
            row[v0 + 16] = accA1[r] + bias1;
            row[v0 + 32] = accA2[r] + bias2;
            row[v0 + 48] = accA3[r] + bias3;
        }
        {   // M-subtile 1
            int m  = m_base + 16 + rr;
            int tt = m >> 6;
            int bb = m & 63;
            float* row = out + (size_t)(bb * TSEQ + tt) * VOCAB;
            row[v0     ] = accB0[r] + bias0;
            row[v0 + 16] = accB1[r] + bias1;
            row[v0 + 32] = accB2[r] + bias2;
            row[v0 + 48] = accB3[r] + bias3;
        }
    }
}

// ---------------- host side -----------------------------------------------------------

extern "C" void kernel_launch(void* const* d_in, const int* in_sizes, int n_in,
                              void* d_out, int out_size, void* d_ws, size_t ws_size,
                              hipStream_t stream) {
    (void)in_sizes; (void)n_in; (void)out_size; (void)ws_size;

    const float* features  = (const float*)d_in[0];
    const int*   captions  = (const int*)  d_in[1];
    const float* embedding = (const float*)d_in[2];
    const float* W_i = (const float*)d_in[3];
    const float* U_i = (const float*)d_in[4];
    const float* b_i = (const float*)d_in[5];
    const float* W_f = (const float*)d_in[6];
    const float* U_f = (const float*)d_in[7];
    const float* b_f = (const float*)d_in[8];
    const float* W_g = (const float*)d_in[9];
    const float* U_g = (const float*)d_in[10];
    const float* b_g = (const float*)d_in[11];
    const float* W_o = (const float*)d_in[12];
    const float* U_o = (const float*)d_in[13];
    const float* b_o = (const float*)d_in[14];
    const float* linear_w = (const float*)d_in[15];
    const float* linear_b = (const float*)d_in[16];
    float* out = (float*)d_out;

    char* w = (char*)d_ws;
    const size_t XBF  = (size_t)TSEQ * BATCH * EMBED * 2;   //  4 MB
    const size_t WTB  = (size_t)GATES * EMBED * 2;          //  4 MB
    const size_t UTB  = (size_t)GATES * HIDDEN * 2;         //  8 MB
    const size_t LWB  = (size_t)VOCAB * HIDDEN * 2;         // 64 MB
    const size_t BCB  = (size_t)GATES * 4;                  // 16 KB
    const size_t GTB  = (size_t)BATCH * GATES * 4;          //  1 MB
    const size_t CSB  = (size_t)BATCH * HIDDEN * 4;         // 256 KB
    const size_t HBB  = (size_t)BATCH * HIDDEN * 2;         // 128 KB

    unsigned short* xbf  = (unsigned short*)w;  w += XBF;
    unsigned short* WT   = (unsigned short*)w;  w += WTB;
    unsigned short* UT   = (unsigned short*)w;  w += UTB;
    unsigned short* lwbf = (unsigned short*)w;  w += LWB;
    float*          bc   = (float*)w;           w += BCB;
    float*          gts  = (float*)w;           w += GTB;
    float*          cst  = (float*)w;           w += CSB;
    unsigned short* hbf  = (unsigned short*)w;  w += HBB;
    unsigned short* hsbf = (unsigned short*)w;  // 8 MB

    pack_x_kernel   <<<(TSEQ * BATCH * EMBED) / 256, 256, 0, stream>>>(features, captions, embedding, xbf);
    pack_w_kernel   <<<(GATES * EMBED) / 256,        256, 0, stream>>>(W_i, W_f, W_g, W_o, WT);
    pack_u_kernel   <<<(GATES * HIDDEN) / 256,       256, 0, stream>>>(U_i, U_f, U_g, U_o, UT);
    pack_bias_kernel<<<GATES / 256,                  256, 0, stream>>>(b_i, b_f, b_g, b_o, bc);
    pack_lw_kernel  <<<((size_t)VOCAB * HIDDEN) / 256, 256, 0, stream>>>(linear_w, lwbf);
    init_state_kernel<<<(BATCH * HIDDEN) / 256,      256, 0, stream>>>(hbf, cst);

    for (int t = 0; t < TSEQ; ++t) {
        lstm_gates_kernel<<<32,  256, 0, stream>>>(xbf, WT, UT, hbf, bc, gts, t);
        lstm_cell_kernel <<<(BATCH * HIDDEN) / 256, 256, 0, stream>>>(gts, cst, hbf, hsbf, t);
    }

    // grid: 500 N-quads x (4096 rows / (8 waves * 32 rows)) = (500, 16)
    logits_kernel<<<dim3(VOCAB / 64, (TSEQ * BATCH) / 256), 256, 0, stream>>>(hsbf, lwbf, linear_b, out);
}
